// Multi_Hetero_24773371363388
// MI455X (gfx1250) — compile-verified, matrix operand order checked by reading, not run on hardware
//
#include <hip/hip_runtime.h>

// ---------------------------------------------------------------------------
// CDNA5 / gfx1250 Multi-Hetero-GNN forward.
// Dense linears: v_wmma_f32_16x16x32_bf16 with weights pre-swizzled once into
// per-lane bf16 B-fragment layout (weights reused by ~6250 row blocks, so the
// column gather + cvt cost is hoisted out of the hot loop entirely).
// Graph aggregation: edge-parallel float4 gathers + global f32 atomics.
// ---------------------------------------------------------------------------

typedef __attribute__((ext_vector_type(16))) __bf16 v16bf;
typedef __attribute__((ext_vector_type(8)))  float  v8f;

#define NEG_SLOPE 0.1f
#define NP_PAPER  100000
#define N_AUTHOR  50000
#define E_PP      800000
#define E_AP      400000
#define E_PA      400000

// ---- A fragment (16x32 bf16): lane = row, half selects K interleave --------
__device__ __forceinline__ v16bf load_a_frag(const float* __restrict__ A,
                                             int row, int kbase, int half) {
  const float* p0 = A + (size_t)row * 128 + kbase + half * 8;
  const float* p1 = p0 + 16;
  float4 a0 = ((const float4*)p0)[0];
  float4 a1 = ((const float4*)p0)[1];
  float4 a2 = ((const float4*)p1)[0];
  float4 a3 = ((const float4*)p1)[1];
  v16bf f;
  f[0]=(__bf16)a0.x;  f[1]=(__bf16)a0.y;  f[2]=(__bf16)a0.z;  f[3]=(__bf16)a0.w;
  f[4]=(__bf16)a1.x;  f[5]=(__bf16)a1.y;  f[6]=(__bf16)a1.z;  f[7]=(__bf16)a1.w;
  f[8]=(__bf16)a2.x;  f[9]=(__bf16)a2.y;  f[10]=(__bf16)a2.z; f[11]=(__bf16)a2.w;
  f[12]=(__bf16)a3.x; f[13]=(__bf16)a3.y; f[14]=(__bf16)a3.z; f[15]=(__bf16)a3.w;
  return f;
}

// ---- Weight pre-swizzle: W[128 x dout] fp32 (+optional W2 summed) ---------
// -> bf16 B-fragments: out[(tile*4 + kt)*32 + lane] = 16 bf16 (32 B, aligned).
// One thread per (tile, kt, lane); total = ntiles * 128 threads.
__global__ void pack_b_kernel(const float* __restrict__ W,
                              const float* __restrict__ W2,
                              int ldW, v16bf* __restrict__ out, int ntiles) {
  int t = blockIdx.x * blockDim.x + threadIdx.x;
  if (t >= ntiles * 128) return;
  int lane = t & 31;
  int kt   = (t >> 5) & 3;
  int tile = t >> 7;
  int half = lane >> 4;
  int col  = tile * 16 + (lane & 15);
  int kbase = kt * 32;
  v16bf f;
#pragma unroll
  for (int e = 0; e < 8; ++e) {
    int k0 = kbase + half * 8 + e;
    int k1 = k0 + 16;
    float w0 = W[(size_t)k0 * ldW + col];
    float w1 = W[(size_t)k1 * ldW + col];
    if (W2) { w0 += W2[(size_t)k0 * ldW + col]; w1 += W2[(size_t)k1 * ldW + col]; }
    f[e]     = (__bf16)w0;
    f[e + 8] = (__bf16)w1;
  }
  out[t] = f;
}

// ---- Full K=128 GEMM term with pre-packed B: 4 x v_wmma_f32_16x16x32_bf16 --
__device__ __forceinline__ v8f mm128p(v8f acc, const float* __restrict__ A,
                                      const v16bf* __restrict__ Bp,
                                      int tile, int r0, int l16, int half,
                                      int lane) {
#pragma unroll
  for (int kt = 0; kt < 4; ++kt) {
    v16bf a = load_a_frag(A, r0 + l16, kt * 32, half);
    v16bf b = Bp[(tile * 4 + kt) * 32 + lane];
    acc = __builtin_amdgcn_wmma_f32_16x16x32_bf16(false, a, false, b,
                                                  (short)0, acc, false, false);
  }
  return acc;
}

// ---- Fused hetero-SAGE conv: Y = act(A0@B0 + [A1@B1] + A2@B2 + bias) -------
// dout = 128, 8 waves/block each owning one 16-col tile; 16 rows/block.
__global__ __launch_bounds__(256) void sage3_wmma(
    const float* __restrict__ A0, const v16bf* __restrict__ B0,
    const float* __restrict__ A1, const v16bf* __restrict__ B1,
    const float* __restrict__ A2, const v16bf* __restrict__ B2,
    const float* __restrict__ b0, const float* __restrict__ b1,
    const float* __restrict__ b2, const float* __restrict__ b3,
    float* __restrict__ Y, int relu) {
  int lane = threadIdx.x & 31;
  int wave = threadIdx.x >> 5;
  int half = lane >> 4;
  int l16  = lane & 15;
  int r0   = blockIdx.x * 16;
  int colg = wave * 16 + l16;

  v8f acc = {};
  acc = mm128p(acc, A0, B0, wave, r0, l16, half, lane);
  if (A1) acc = mm128p(acc, A1, B1, wave, r0, l16, half, lane);
  acc = mm128p(acc, A2, B2, wave, r0, l16, half, lane);

  float bs = b0[colg] + b1[colg];
  if (b2) bs += b2[colg];
  if (b3) bs += b3[colg];

#pragma unroll
  for (int v = 0; v < 8; ++v) {
    float x = acc[v] + bs;
    if (relu) x = (x > 0.0f) ? x : NEG_SLOPE * x;
    Y[(size_t)(r0 + half * 8 + v) * 128 + colg] = x;
  }
}

// ---- Generic single-term GEMM (K=128, dout = blockDim.x/32*16) -------------
__global__ __launch_bounds__(128) void gemm_wmma(
    const float* __restrict__ A, const v16bf* __restrict__ Bp,
    const float* __restrict__ bias, float* __restrict__ Y,
    int ldY, int colOff) {
  int lane = threadIdx.x & 31;
  int wave = threadIdx.x >> 5;
  int half = lane >> 4;
  int l16  = lane & 15;
  int r0   = blockIdx.x * 16;
  int col  = wave * 16 + l16;

  v8f acc = {};
  acc = mm128p(acc, A, Bp, wave, r0, l16, half, lane);
  float bs = bias[col];
#pragma unroll
  for (int v = 0; v < 8; ++v)
    Y[(size_t)(r0 + half * 8 + v) * ldY + colOff + col] = acc[v] + bs;
}

// ---- Graph aggregation ----------------------------------------------------
__global__ void degree_kernel(const int* __restrict__ dst, int E,
                              float* __restrict__ cnt) {
  int t = blockIdx.x * blockDim.x + threadIdx.x;
  if (t < E) atomicAdd(&cnt[dst[t]], 1.0f);
}

// one wave per edge; lane handles 4 consecutive floats (float4 gather)
__global__ void scatter_add_kernel(const float* __restrict__ X,
                                   const int* __restrict__ src,
                                   const int* __restrict__ dst, int E,
                                   float* __restrict__ agg) {
  int t = blockIdx.x * blockDim.x + threadIdx.x;
  int e = t >> 5, lane = t & 31;
  if (e >= E) return;
  int s = src[e], d = dst[e];
  float4 v = ((const float4*)(X + (size_t)s * 128))[lane];
  float* out = agg + (size_t)d * 128 + lane * 4;
  atomicAdd(out + 0, v.x);
  atomicAdd(out + 1, v.y);
  atomicAdd(out + 2, v.z);
  atomicAdd(out + 3, v.w);
}

__global__ void norm_kernel(float* __restrict__ agg,
                            const float* __restrict__ cnt, int M) {
  int t = blockIdx.x * blockDim.x + threadIdx.x;
  if (t >= M * 32) return;
  float inv = 1.0f / fmaxf(cnt[t >> 5], 1.0f);
  float4* p = ((float4*)agg) + t;
  float4 v = *p;
  v.x *= inv; v.y *= inv; v.z *= inv; v.w *= inv;
  *p = v;
}

// ---- Attention combine: att = minmax-norm(ga@Wagg+b - mean); weighted sum --
__global__ __launch_bounds__(256) void att_combine_kernel(
    const float* __restrict__ GA, const float* __restrict__ Wagg,
    const float* __restrict__ bagg,
    const float* __restrict__ E0, const float* __restrict__ E1,
    const float* __restrict__ E2, float* __restrict__ F, int N) {
  int node = blockIdx.x * 8 + (threadIdx.x >> 5);
  int lane = threadIdx.x & 31;
  if (node >= N) return;
  const float* ga = GA + (size_t)node * 192;
  float att[3];
#pragma unroll
  for (int j = 0; j < 3; ++j) {
    float p = 0.0f;
    for (int k = lane; k < 192; k += 32) p += ga[k] * Wagg[k * 3 + j];
#pragma unroll
    for (int off = 16; off > 0; off >>= 1) p += __shfl_xor(p, off, 32);
    att[j] = p + bagg[j];
  }
  float mean = (att[0] + att[1] + att[2]) * (1.0f / 3.0f);
  att[0] -= mean; att[1] -= mean; att[2] -= mean;
  float mn = fminf(att[0], fminf(att[1], att[2]));
  float mx = fmaxf(att[0], fmaxf(att[1], att[2]));
  float inv = 1.0f / (mx - mn);
  float w0 = (att[0] - mn) * inv + (1.0f / 3.0f);
  float w1 = (att[1] - mn) * inv + (1.0f / 3.0f);
  float w2 = (att[2] - mn) * inv + (1.0f / 3.0f);
  float4 a = ((const float4*)(E0 + (size_t)node * 128))[lane];
  float4 b = ((const float4*)(E1 + (size_t)node * 128))[lane];
  float4 c = ((const float4*)(E2 + (size_t)node * 128))[lane];
  float4 r;
  r.x = w0 * a.x + w1 * b.x + w2 * c.x;
  r.y = w0 * a.y + w1 * b.y + w2 * c.y;
  r.z = w0 * a.z + w1 * b.z + w2 * c.z;
  r.w = w0 * a.w + w1 * b.w + w2 * c.w;
  ((float4*)(F + (size_t)node * 128))[lane] = r;
}

// ---------------------------------------------------------------------------
extern "C" void kernel_launch(void* const* d_in, const int* in_sizes, int n_in,
                              void* d_out, int out_size, void* d_ws, size_t ws_size,
                              hipStream_t stream) {
  (void)in_sizes; (void)n_in; (void)out_size; (void)ws_size;

  const float* xp = (const float*)d_in[0];
  const float* xa = (const float*)d_in[1];
  const int* src_pp = (const int*)d_in[2]; const int* dst_pp = src_pp + E_PP;
  const int* src_ap = (const int*)d_in[3]; const int* dst_ap = src_ap + E_AP;
  const int* src_pa = (const int*)d_in[4]; const int* dst_pa = src_pa + E_PA;
  const float* aggW = (const float*)d_in[5];
  const float* aggb = (const float*)d_in[6];
  const float* attW[3] = {(const float*)d_in[7], (const float*)d_in[9], (const float*)d_in[11]};
  const float* attb[3] = {(const float*)d_in[8], (const float*)d_in[10], (const float*)d_in[12]};
  const float* decW = (const float*)d_in[13];
  const float* decb = (const float*)d_in[14];
  // model weights: per (m, l): ap.lW, ap.lb, ap.rW, ap.rb, pa.*, pp.*
  auto MW = [&](int m, int l, int idx) -> const float* {
    return (const float*)d_in[15 + (m * 3 + l) * 12 + idx];
  };

  // workspace carve-out
  char* ws = (char*)d_ws;
  size_t off = 0;
  auto carve = [&](size_t bytes) -> void* {
    void* p = ws + off;
    off = (off + bytes + 255) & ~(size_t)255;
    return p;
  };
  const size_t PB = (size_t)NP_PAPER * 128 * 4;   // paper feature matrix
  const size_t AB = (size_t)N_AUTHOR * 128 * 4;   // author feature matrix
  const size_t WPK = 8 * 4 * 32 * sizeof(v16bf);  // packed 128x128 weight (32KB)
  float* AGG_PP = (float*)carve(PB);
  float* AGG_AP = (float*)carve(PB);
  float* AGG_PA = (float*)carve(AB);
  float* CNT_PP = (float*)carve((size_t)NP_PAPER * 4);
  float* CNT_AP = (float*)carve((size_t)NP_PAPER * 4);
  float* CNT_PA = (float*)carve((size_t)N_AUTHOR * 4);
  float* HP_A = (float*)carve(PB);
  float* HP_B = (float*)carve(PB);
  float* HA_A = (float*)carve(AB);
  float* HA_B = (float*)carve(AB);
  float* EMB[3] = {(float*)carve(PB), (float*)carve(PB), (float*)carve(PB)};
  float* GA    = (float*)carve((size_t)NP_PAPER * 192 * 4);
  float* FINAL = (float*)carve(PB);
  v16bf* PK0 = (v16bf*)carve(WPK);
  v16bf* PK1 = (v16bf*)carve(WPK);
  v16bf* PK2 = (v16bf*)carve(WPK);

  // --- degrees (feature independent: compute once, reuse for all 9 convs) ---
  hipMemsetAsync(CNT_PP, 0, (size_t)NP_PAPER * 4, stream);
  hipMemsetAsync(CNT_AP, 0, (size_t)NP_PAPER * 4, stream);
  hipMemsetAsync(CNT_PA, 0, (size_t)N_AUTHOR * 4, stream);
  degree_kernel<<<(E_PP + 255) / 256, 256, 0, stream>>>(dst_pp, E_PP, CNT_PP);
  degree_kernel<<<(E_AP + 255) / 256, 256, 0, stream>>>(dst_ap, E_AP, CNT_AP);
  degree_kernel<<<(E_PA + 255) / 256, 256, 0, stream>>>(dst_pa, E_PA, CNT_PA);

  const int PBLK = NP_PAPER / 16;  // 6250
  const int ABLK = N_AUTHOR / 16;  // 3125

  for (int m = 0; m < 3; ++m) {
    for (int l = 0; l < 3; ++l) {
      const float* xpi = (l == 0) ? xp : (l == 1 ? HP_A : HP_B);
      const float* xai = (l == 0) ? xa : (l == 1 ? HA_A : HA_B);
      bool last = (l == 2);

      // mean-aggregate per edge type
      hipMemsetAsync(AGG_PP, 0, PB, stream);
      scatter_add_kernel<<<(int)(((size_t)E_PP * 32 + 255) / 256), 256, 0, stream>>>(
          xpi, src_pp, dst_pp, E_PP, AGG_PP);
      norm_kernel<<<(NP_PAPER * 32 + 255) / 256, 256, 0, stream>>>(AGG_PP, CNT_PP, NP_PAPER);

      hipMemsetAsync(AGG_AP, 0, PB, stream);
      scatter_add_kernel<<<(int)(((size_t)E_AP * 32 + 255) / 256), 256, 0, stream>>>(
          xai, src_ap, dst_ap, E_AP, AGG_AP);
      norm_kernel<<<(NP_PAPER * 32 + 255) / 256, 256, 0, stream>>>(AGG_AP, CNT_AP, NP_PAPER);

      if (!last) {  // author output unused after last conv
        hipMemsetAsync(AGG_PA, 0, AB, stream);
        scatter_add_kernel<<<(int)(((size_t)E_PA * 32 + 255) / 256), 256, 0, stream>>>(
            xpi, src_pa, dst_pa, E_PA, AGG_PA);
        norm_kernel<<<(N_AUTHOR * 32 + 255) / 256, 256, 0, stream>>>(AGG_PA, CNT_PA, N_AUTHOR);
      }

      // pre-swizzle this conv's paper-side weights into bf16 B-fragments
      pack_b_kernel<<<4, 256, 0, stream>>>(MW(m, l, 8), nullptr, 128, PK0, 8);      // pp.l
      pack_b_kernel<<<4, 256, 0, stream>>>(MW(m, l, 0), nullptr, 128, PK1, 8);      // ap.l
      pack_b_kernel<<<4, 256, 0, stream>>>(MW(m, l, 10), MW(m, l, 2), 128, PK2, 8); // pp.r+ap.r

      // paper out: agg_pp@Wl_pp + agg_ap@Wl_ap + xp@(Wr_pp+Wr_ap) + biases
      float* Yp = last ? EMB[m] : (l == 0 ? HP_A : HP_B);
      sage3_wmma<<<PBLK, 256, 0, stream>>>(
          AGG_PP, PK0, AGG_AP, PK1, xpi, PK2,
          MW(m, l, 9), MW(m, l, 11), MW(m, l, 1), MW(m, l, 3),
          Yp, last ? 0 : 1);

      if (!last) {  // author out: agg_pa@Wl_pa + xa@Wr_pa
        pack_b_kernel<<<4, 256, 0, stream>>>(MW(m, l, 4), nullptr, 128, PK0, 8); // pa.l
        pack_b_kernel<<<4, 256, 0, stream>>>(MW(m, l, 6), nullptr, 128, PK1, 8); // pa.r
        float* Ya = (l == 0) ? HA_A : HA_B;
        sage3_wmma<<<ABLK, 256, 0, stream>>>(
            AGG_PA, PK0, nullptr, nullptr, xai, PK1,
            MW(m, l, 5), MW(m, l, 7), nullptr, nullptr,
            Ya, 1);
      }
    }
  }

  // --- attention: ga_i = EMB[i] @ attW[i] + attb[i] into GA[:, i*64:(i+1)*64]
  for (int i = 0; i < 3; ++i) {
    pack_b_kernel<<<2, 256, 0, stream>>>(attW[i], nullptr, 64, PK0, 4);
    gemm_wmma<<<PBLK, 128, 0, stream>>>(EMB[i], PK0, attb[i], GA,
                                        /*ldY=*/192, /*colOff=*/i * 64);
  }

  att_combine_kernel<<<(NP_PAPER + 7) / 8, 256, 0, stream>>>(
      GA, aggW, aggb, EMB[0], EMB[1], EMB[2], FINAL, NP_PAPER);

  // --- decode: out = FINAL @ decW + decb  -> [100000, 64]
  pack_b_kernel<<<2, 256, 0, stream>>>(decW, nullptr, 64, PK0, 4);
  gemm_wmma<<<PBLK, 128, 0, stream>>>(FINAL, PK0, decb, (float*)d_out,
                                      /*ldY=*/64, /*colOff=*/0);
}
